// GIN_10453950399025
// MI455X (gfx1250) — matrix operand order repaired and checked by reading, not compiled, not run on hardware
//
#include <hip/hip_runtime.h>
#include <hip/hip_bf16.h>

// ---------------- types ----------------
typedef __bf16 bf16;
typedef __attribute__((ext_vector_type(16))) __bf16 v16bf;
typedef __attribute__((ext_vector_type(8)))  __bf16 v8bf;
typedef __attribute__((ext_vector_type(8)))  float  v8f;

#define N_NODES   50000
#define M_PAD     50016              // padded to 32-row GEMM wave tiles (1563 tiles)
#define N_EDGES   400000
#define N_GRAPHS  250
#define DD        256
#define NUM_CLASSES 10
#define BN_EPS    1e-5f

// native f32 atomic add (global_atomic_add_f32), never a CAS loop
__device__ __forceinline__ void atomAddF32(float* p, float v) { unsafeAtomicAdd(p, v); }

// ---------------- utility kernels ----------------
__global__ void k_zero(float* __restrict__ p, long n) {
  long i = (long)blockIdx.x * blockDim.x + threadIdx.x;
  if (i < n) p[i] = 0.0f;
}

__global__ void k_copy(const float* __restrict__ s, float* __restrict__ d, long n) {
  long i = (long)blockIdx.x * blockDim.x + threadIdx.x;
  if (i < n) d[i] = s[i];
}

__global__ void k_pack_bf16(const float* __restrict__ W, bf16* __restrict__ Wb, int n) {
  int i = blockIdx.x * blockDim.x + threadIdx.x;
  if (i < n) Wb[i] = (bf16)W[i];
}

// ---------------- graph aggregation ----------------
__global__ void k_deg(const int* __restrict__ dst, float* __restrict__ deg) {
  int e = blockIdx.x * blockDim.x + threadIdx.x;
  if (e < N_EDGES) atomAddF32(&deg[dst[e]], 1.0f);
}

// one thread = (edge, 4-float chunk); 400000 * 64 threads
__global__ void k_agg(const float* __restrict__ h, const int* __restrict__ src,
                      const int* __restrict__ dst, float* __restrict__ agg) {
  long t = (long)blockIdx.x * blockDim.x + threadIdx.x;
  int e = (int)(t >> 6);
  int c = ((int)t & 63) << 2;
  if (e < N_EDGES) {
    const float4 v = *(const float4*)(h + (long)src[e] * DD + c);
    float* o = agg + (long)dst[e] * DD + c;
    atomAddF32(o + 0, v.x);
    atomAddF32(o + 1, v.y);
    atomAddF32(o + 2, v.z);
    atomAddF32(o + 3, v.w);
  }
}

// h_pre = h + mean_agg, emitted as bf16 GEMM input; zero-fills padded rows
__global__ void k_hpre(const float* __restrict__ h, const float* __restrict__ agg,
                       const float* __restrict__ deg, bf16* __restrict__ abf) {
  long i = (long)blockIdx.x * blockDim.x + threadIdx.x;
  if (i < (long)M_PAD * DD) {
    int row = (int)(i >> 8);
    if (row >= N_NODES) { abf[i] = (bf16)0.0f; return; }
    float d = deg[row];
    float m = (d > 0.0f) ? agg[i] / fmaxf(d, 1.0f) : 0.0f;
    abf[i] = (bf16)(h[i] + m);
  }
}

// ---------------- WMMA GEMM: [M_PAD x 256](bf16) x [256 x 256](bf16) + bias, relu ----------------
// Per wave: 32 rows x 64 cols = 2 M-tiles x 4 N-tiles, K loop 8 x 32.
// A fragment (16-bit A 16x32 layout): lane = row (lane&15), two 16B runs at
// kc+8*(lane>>4) and kc+16+8*(lane>>4). B fragment: lane = B column (lane&15),
// same k order as A (gathered). Each B fragment feeds two WMMAs.
__device__ __forceinline__ v16bf load_afrag(const bf16* arow, int kc, int half) {
  v8bf lo = *(const v8bf*)(arow + kc + 8 * half);
  v8bf hi = *(const v8bf*)(arow + kc + 16 + 8 * half);
  return __builtin_shufflevector(lo, hi,
      0, 1, 2, 3, 4, 5, 6, 7, 8, 9, 10, 11, 12, 13, 14, 15);
}

__global__ __launch_bounds__(256) void k_gemm(const bf16* __restrict__ A,
                                              const bf16* __restrict__ W,
                                              const float* __restrict__ bias,
                                              float* __restrict__ out_f32,
                                              bf16* __restrict__ out_bf16,
                                              int do_relu) {
  const int TILES_M = M_PAD / 32;              // 1563
  const int lane = threadIdx.x & 31;
  const int wid  = blockIdx.x * (blockDim.x >> 5) + (threadIdx.x >> 5);
  if (wid >= TILES_M * 4) return;              // whole waves retire uniformly (EXEC stays all-1)

  const int nq = wid / TILES_M;                // 0..3: 64-column strip
  const int mq = wid % TILES_M;
  const int half = lane >> 4;
  const int lm   = lane & 15;
  const int m0 = mq * 32;
  const int n_base = nq * 64;

  const bf16* arow0 = A + (long)(m0 + lm) * DD;
  const bf16* arow1 = arow0 + 16 * DD;

  v8f zero = {0.f, 0.f, 0.f, 0.f, 0.f, 0.f, 0.f, 0.f};
  v8f acc[2][4];
#pragma unroll
  for (int mt = 0; mt < 2; mt++)
#pragma unroll
    for (int nt = 0; nt < 4; nt++) acc[mt][nt] = zero;

  for (int kc = 0; kc < DD; kc += 32) {
    v16bf a0 = load_afrag(arow0, kc, half);
    v16bf a1 = load_afrag(arow1, kc, half);
    const bf16* bcol = W + (long)kc * DD + n_base + lm;
#pragma unroll
    for (int nt = 0; nt < 4; nt++) {
      v16bf b;
#pragma unroll
      for (int s = 0; s < 16; s++) {
        int kk = (s < 8) ? (8 * half + s) : (16 + 8 * half + (s - 8));
        b[s] = bcol[(long)kk * DD + nt * 16];
      }
      acc[0][nt] = __builtin_amdgcn_wmma_f32_16x16x32_bf16(
          false, a0, false, b, (short)0, acc[0][nt], false, false);
      acc[1][nt] = __builtin_amdgcn_wmma_f32_16x16x32_bf16(
          false, a1, false, b, (short)0, acc[1][nt], false, false);
    }
  }

  // C/D layout: VGPR r -> row base + 8*half + r, column n_base + nt*16 + lm
#pragma unroll
  for (int nt = 0; nt < 4; nt++) {
    int n = n_base + nt * 16 + lm;
    float bv = bias[n];
#pragma unroll
    for (int mt = 0; mt < 2; mt++) {
      int rbase = m0 + mt * 16 + half * 8;
#pragma unroll
      for (int r = 0; r < 8; r++) {
        float v = acc[mt][nt][r] + bv;
        if (do_relu) v = fmaxf(v, 0.0f);
        long o = (long)(rbase + r) * DD + n;
        if (out_f32) out_f32[o] = v;
        else         out_bf16[o] = (bf16)v;
      }
    }
  }
}

// ---------------- batch norm ----------------
__global__ void k_bnstats(const float* __restrict__ h2, float* __restrict__ colsum,
                          float* __restrict__ colsq) {
  int j = threadIdx.x;                 // column
  float s = 0.0f, q = 0.0f;
  for (int r = blockIdx.x; r < N_NODES; r += gridDim.x) {
    float v = h2[(long)r * DD + j];
    s += v; q += v * v;
  }
  atomAddF32(&colsum[j], s);
  atomAddF32(&colsq[j], q);
}

__global__ void k_bnfinal(const float* __restrict__ colsum, const float* __restrict__ colsq,
                          const float* __restrict__ gamma, const float* __restrict__ beta,
                          float* __restrict__ scale, float* __restrict__ shift) {
  int j = threadIdx.x;
  float inv_n = 1.0f / (float)N_NODES;
  float mu  = colsum[j] * inv_n;
  float var = colsq[j] * inv_n - mu * mu;
  float sc  = gamma[j] * rsqrtf(var + BN_EPS);
  scale[j] = sc;
  shift[j] = beta[j] - mu * sc;
}

__global__ void k_bnapply(const float* __restrict__ h2, const float* __restrict__ scale,
                          const float* __restrict__ shift, float* __restrict__ h) {
  long i = (long)blockIdx.x * blockDim.x + threadIdx.x;
  if (i < (long)N_NODES * DD) {
    int j = (int)(i & 255);
    h[i] = h2[i] * scale[j] + shift[j];
  }
}

// ---------------- readout ----------------
__global__ void k_lastidx(const int* __restrict__ counts, int* __restrict__ idx) {
  if (threadIdx.x == 0 && blockIdx.x == 0) {
    int c = 0;
    for (int g = 0; g < N_GRAPHS; g++) { c += counts[g]; idx[g] = c - 1; }
  }
}

__global__ void k_fc1(const float* __restrict__ h, const int* __restrict__ idx,
                      const float* __restrict__ W, const float* __restrict__ b,
                      float* __restrict__ g1) {
  __shared__ float row[DD];
  int j = threadIdx.x;
  int g = blockIdx.x;
  row[j] = h[(long)idx[g] * DD + j];
  __syncthreads();
  float acc = b[j];
  for (int k = 0; k < DD; k++) acc += row[k] * W[(long)k * DD + j];
  g1[(long)g * DD + j] = fmaxf(acc, 0.0f);
}

__global__ void k_logits(const float* __restrict__ g1, const float* __restrict__ W,
                         const float* __restrict__ b, float* __restrict__ out) {
  int g = blockIdx.x * blockDim.x + threadIdx.x;
  if (g >= N_GRAPHS) return;
  float lg[NUM_CLASSES];
#pragma unroll
  for (int c = 0; c < NUM_CLASSES; c++) lg[c] = b[c];
  for (int k = 0; k < DD; k++) {
    float x = g1[(long)g * DD + k];
#pragma unroll
    for (int c = 0; c < NUM_CLASSES; c++) lg[c] += x * W[k * NUM_CLASSES + c];
  }
  float m = lg[0];
#pragma unroll
  for (int c = 1; c < NUM_CLASSES; c++) m = fmaxf(m, lg[c]);
  float s = 0.0f;
#pragma unroll
  for (int c = 0; c < NUM_CLASSES; c++) s += expf(lg[c] - m);
  float lse = m + logf(s);
#pragma unroll
  for (int c = 0; c < NUM_CLASSES; c++) out[g * NUM_CLASSES + c] = lg[c] - lse;
}

// ---------------- host driver ----------------
extern "C" void kernel_launch(void* const* d_in, const int* in_sizes, int n_in,
                              void* d_out, int out_size, void* d_ws, size_t ws_size,
                              hipStream_t stream) {
  (void)in_sizes; (void)n_in; (void)out_size; (void)ws_size;

  const float* x     = (const float*)d_in[0];
  const int*   src   = (const int*)d_in[1];
  const int*   dst   = (const int*)d_in[2];
  const int*   ncnt  = (const int*)d_in[3];
  const float* W1a   = (const float*)d_in[4];
  const float* b1a   = (const float*)d_in[5];
  const float* W1b   = (const float*)d_in[6];
  const float* b1b   = (const float*)d_in[7];
  const float* g1w   = (const float*)d_in[8];
  const float* be1   = (const float*)d_in[9];
  const float* Ws1   = (const float*)d_in[10];
  const float* bs1   = (const float*)d_in[11];
  const float* Ws2   = (const float*)d_in[12];
  const float* bs2   = (const float*)d_in[13];
  const float* gms   = (const float*)d_in[14];
  const float* bts   = (const float*)d_in[15];
  const float* fcW1  = (const float*)d_in[16];
  const float* fcb1  = (const float*)d_in[17];
  const float* fcW2  = (const float*)d_in[18];
  const float* fcb2  = (const float*)d_in[19];
  float* out = (float*)d_out;

  const long ND     = (long)N_NODES * DD;      // 12.8M
  const long ND_PAD = (long)M_PAD * DD;        // padded activation size
  char* ws = (char*)d_ws;
  size_t off = 0;
  auto carve = [&](size_t bytes) -> char* {
    char* p = ws + off;
    off = (off + bytes + 255) & ~(size_t)255;
    return p;
  };
  float* h      = (float*)carve(ND * 4);       // layer input (f32)
  float* agg    = (float*)carve(ND_PAD * 4);   // scatter accum; reused as h2 (padded)
  float* h2     = agg;                         // MLP output pre-BN (aliases agg)
  bf16*  abf    = (bf16*)carve(ND_PAD * 2);    // bf16 GEMM-A input (padded)
  bf16*  tbf    = (bf16*)carve(ND_PAD * 2);    // bf16 MLP hidden (padded)
  bf16*  wbf    = (bf16*)carve((size_t)10 * DD * DD * 2);   // packed weights
  float* deg    = (float*)carve((size_t)N_NODES * 4);
  float* colsum = (float*)carve(DD * 4);       // colsum + colsq contiguous
  float* colsq  = (float*)carve(DD * 4);
  float* scale  = (float*)carve(DD * 4);
  float* shift  = (float*)carve(DD * 4);
  int*   lidx   = (int*)carve((size_t)N_GRAPHS * 4);
  float* g1buf  = (float*)carve((size_t)N_GRAPHS * DD * 4);

  const int TB = 256;
  const int ZB_ND   = (int)((ND + TB - 1) / TB);                 // 50000
  const int ZB_PAD  = (int)((ND_PAD + TB - 1) / TB);             // 50016
  const int GEMM_B  = (M_PAD / 32 * 4 + 7) / 8;                  // 782 blocks x 8 waves
  const int AGG_B   = (int)(((long)N_EDGES * 64 + TB - 1) / TB); // 100000
  const int WSZ     = DD * DD;                                   // 65536

  // degree (constant across layers)
  k_zero<<<(N_NODES + TB - 1) / TB, TB, 0, stream>>>(deg, N_NODES);
  k_deg<<<(N_EDGES + TB - 1) / TB, TB, 0, stream>>>(dst, deg);

  // pack weights to bf16: slot0=W1a, slot1=W1b, slot2..5=Ws1[i], slot6..9=Ws2[i]
  k_pack_bf16<<<WSZ / TB, TB, 0, stream>>>(W1a, wbf + 0 * WSZ, WSZ);
  k_pack_bf16<<<WSZ / TB, TB, 0, stream>>>(W1b, wbf + 1 * WSZ, WSZ);
  for (int i = 0; i < 4; i++) {
    k_pack_bf16<<<WSZ / TB, TB, 0, stream>>>(Ws1 + (long)i * WSZ, wbf + (2 + i) * WSZ, WSZ);
    k_pack_bf16<<<WSZ / TB, TB, 0, stream>>>(Ws2 + (long)i * WSZ, wbf + (6 + i) * WSZ, WSZ);
  }

  // h = x
  k_copy<<<ZB_ND, TB, 0, stream>>>(x, h, ND);

  for (int L = 0; L < 5; L++) {
    const bf16 *w1, *w2;
    const float *bb1, *bb2, *gm, *bt;
    if (L == 0) {
      w1 = wbf + 0 * WSZ; w2 = wbf + 1 * WSZ;
      bb1 = b1a; bb2 = b1b; gm = g1w; bt = be1;
    } else {
      int i = L - 1;
      w1 = wbf + (2 + i) * WSZ; w2 = wbf + (6 + i) * WSZ;
      bb1 = bs1 + (long)i * DD; bb2 = bs2 + (long)i * DD;
      gm = gms + (long)i * DD;  bt = bts + (long)i * DD;
    }
    // mean aggregation + residual -> bf16 input (pad rows zeroed)
    k_zero<<<ZB_ND, TB, 0, stream>>>(agg, ND);
    k_agg<<<AGG_B, TB, 0, stream>>>(h, src, dst, agg);
    k_hpre<<<ZB_PAD, TB, 0, stream>>>(h, agg, deg, abf);
    // MLP via WMMA (bf16 in, f32 accumulate)
    k_gemm<<<GEMM_B, TB, 0, stream>>>(abf, w1, bb1, nullptr, tbf, 1);   // relu -> bf16
    k_gemm<<<GEMM_B, TB, 0, stream>>>(tbf, w2, bb2, h2, nullptr, 1);    // relu -> f32
    // batch norm over nodes (pad rows excluded)
    k_zero<<<(2 * DD + TB - 1) / TB, TB, 0, stream>>>(colsum, 2 * DD);
    k_bnstats<<<256, TB, 0, stream>>>(h2, colsum, colsq);
    k_bnfinal<<<1, TB, 0, stream>>>(colsum, colsq, gm, bt, scale, shift);
    k_bnapply<<<ZB_ND, TB, 0, stream>>>(h2, scale, shift, h);
  }

  // readout
  k_lastidx<<<1, 32, 0, stream>>>(ncnt, lidx);
  k_fc1<<<N_GRAPHS, TB, 0, stream>>>(h, lidx, fcW1, fcb1, g1buf);
  k_logits<<<1, TB, 0, stream>>>(g1buf, fcW2, fcb2, out);
}
